// HeteroGNN_66245575573891
// MI455X (gfx1250) — matrix operand order repaired and checked by reading, not compiled, not run on hardware
//
// HeteroGNN forward for MI455X (gfx1250, wave32, WMMA).
//
// Strategy: all 4-layer MLPs (encoder, both GN blocks' edge/node MLPs, decoder)
// run through one generic fused WMMA kernel:
//   - 256 threads = 8 wave32s per workgroup, 128 rows per workgroup
//   - layer 0: A fragments gathered straight from global (segment list handles
//     the [e, x_src, x_dst, u] concats incl. the 256-wide skip concat, so the
//     concat is never materialized); B weight K-panels are staged into LDS
//     (double buffered, 8KB each) so all 8 waves share one set of global loads
//   - layers 1..3: full 32KB weight staged in LDS once per layer; activations
//     ping-pong through 64KB LDS
//   - v_wmma_f32_16x16x32_f16 (f16 in, f32 accumulate) everywhere
//   - final layer fuses the segment-sum scatter via global_atomic_add_f32
// Small VALU kernels handle counts, segment-mean finalize, column means,
// the 1-row global MLPs, and the 4/1-wide edge output heads.
//
// Roofline: ~150 GFLOP of GEMM, ~300MB activation traffic -> both well under
// 100us on 23.3 TB/s HBM + WMMA matrix rates. LDS weight staging removes the
// 8x per-wave redundancy in weight reads (~2.6GB of L2 traffic on the K0=1024
// block-1 edge GEMM alone).

#include <hip/hip_runtime.h>
#include <hip/hip_bf16.h>
#include <initializer_list>
#include <stdint.h>

typedef __attribute__((ext_vector_type(16))) _Float16 v16h;
typedef __attribute__((ext_vector_type(8)))  _Float16 v8h;
typedef __attribute__((ext_vector_type(8)))  float    v8f;

struct SegDesc {
  const _Float16* base;
  const int* idx;
  int stride;   // row stride in elements
  int start;    // start column in padded concat space
  int vw;       // valid width
  int wpad;     // padded width (multiple of 32)
  int mode;     // 0 = identity row, 1 = gather idx[row], 2 = broadcast row 0
};
struct Segs { SegDesc s[8]; int n; };

// ---------------------------------------------------------------------------
// Generic fused 4-layer MLP: [concat segments](K0) ->128->128->128 (ReLU between)
// ---------------------------------------------------------------------------
__global__ __launch_bounds__(256) void k_fused_mlp4(
    int R, Segs segs, int K0,
    const _Float16* __restrict__ Wt0, const float* __restrict__ B0,
    const _Float16* __restrict__ Wt1, const float* __restrict__ B1,
    const _Float16* __restrict__ Wt2, const float* __restrict__ B2,
    const _Float16* __restrict__ Wt3, const float* __restrict__ B3,
    _Float16* __restrict__ out_h, float* __restrict__ out_f,
    float* __restrict__ agg_sum, const int* __restrict__ agg_dst)
{
  // 96KB LDS: act0 | act1 | staged weights. Layer-0 B K-panels alias act1
  // (dead during layer 0).
  __shared__ _Float16 smem[3 * 128 * 128];
  _Float16* act0 = smem;            // 32KB activation ping
  _Float16* act1 = smem + 16384;    // 32KB activation pong
  _Float16* bw   = smem + 32768;    // 32KB staged 128x128 layer weight
  _Float16* pan0 = act1;            // 8KB layer-0 B panel (buffer 0)
  _Float16* pan1 = act1 + 4096;     // 8KB layer-0 B panel (buffer 1)

  const int tid  = threadIdx.x;
  const int lane = tid & 31;
  const int wv   = tid >> 5;
  const int hi   = lane >> 4;      // half-wave select (A: K-offset 0/8, B: K-offset 0/8)
  const int ln   = lane & 15;
  const int m0   = wv * 16;        // this wave's 16-row strip

  const long rb = (long)blockIdx.x * 128;
  const int  arow = m0 + ln;       // row within block this lane fetches for A frags
  long g_arow = rb + arow;
  if (g_arow > (long)R - 1) g_arow = (long)R - 1;

  // cooperative weight staging coordinates: thread -> (n row, 16-half offset)
  const int sn   = tid >> 1;        // 0..127
  const int soff = (tid & 1) * 16;  // 0 or 16

  // Resolve per-segment row base pointers for this lane's A row (layer 0).
  const _Float16* rowp[8];
#pragma unroll
  for (int j = 0; j < 8; ++j) {
    const SegDesc sd = segs.s[j];
    if (j < segs.n) {
      long ri;
      if (sd.mode == 1)      ri = sd.idx[g_arow];
      else if (sd.mode == 2) ri = 0;
      else                   ri = g_arow;
      rowp[j] = sd.base + ri * (long)sd.stride;
    } else {
      rowp[j] = nullptr;
    }
  }

  const v8f zero8 = {0.f, 0.f, 0.f, 0.f, 0.f, 0.f, 0.f, 0.f};
  v8f acc[8];
#pragma unroll
  for (int t = 0; t < 8; ++t) acc[t] = zero8;

  // -------------------- layer 0: A gathered from global, B panels via LDS ----
  // preload panel for chunk 0 (32 x 128 weights, transposed [n][k] layout)
  *(v16h*)(pan0 + sn * 32 + soff) = *(const v16h*)(Wt0 + (size_t)sn * K0 + soff);
  __syncthreads();

  int pb = 0;
  for (int kk = 0; kk < K0; kk += 32, pb ^= 1) {
    _Float16* cur = pb ? pan1 : pan0;
    _Float16* nxt = pb ? pan0 : pan1;
    if (kk + 32 < K0) {
      *(v16h*)(nxt + sn * 32 + soff) =
          *(const v16h*)(Wt0 + (size_t)sn * K0 + kk + 32 + soff);
    }

    v16h a;
#pragma unroll
    for (int i = 0; i < 16; ++i) a[i] = (_Float16)0.f;

#pragma unroll
    for (int j = 0; j < 8; ++j) {
      if (j < segs.n) {
        const SegDesc sd = segs.s[j];
        if (kk >= sd.start && kk < sd.start + sd.wpad) {
          const int kl = (kk - sd.start) + hi * 8;
          const _Float16* p = rowp[j];
          if (sd.vw == sd.wpad) {
            // fast path: two contiguous 16B loads per A fragment
            v8h lo = *(const v8h*)(p + kl);
            v8h h2 = *(const v8h*)(p + kl + 16);
#pragma unroll
            for (int i = 0; i < 8; ++i) { a[i] = lo[i]; a[8 + i] = h2[i]; }
          } else {
            // slow path (encoder: width 8/16 padded to 32) with zero fill
#pragma unroll
            for (int i = 0; i < 8; ++i) {
              int k1 = kl + i, k2 = kl + 16 + i;
              if (k1 < sd.vw) a[i]     = p[k1];
              if (k2 < sd.vw) a[8 + i] = p[k2];
            }
          }
        }
      }
    }

#pragma unroll
    for (int nt = 0; nt < 8; ++nt) {
      const _Float16* bp = cur + (nt * 16 + ln) * 32 + hi * 8;
      v8h b0 = *(const v8h*)(bp);
      v8h b1 = *(const v8h*)(bp + 16);
      v16h b;
#pragma unroll
      for (int i = 0; i < 8; ++i) { b[i] = b0[i]; b[8 + i] = b1[i]; }
      acc[nt] = __builtin_amdgcn_wmma_f32_16x16x32_f16(
          false, a, false, b, (short)0, acc[nt], false, false);
    }
    __syncthreads();   // cur fully consumed; nxt fully staged
  }

  // epilogue layer 0 -> act0 (bias + ReLU, f32 -> f16)
#pragma unroll
  for (int nt = 0; nt < 8; ++nt) {
    float bv = B0[nt * 16 + ln];
#pragma unroll
    for (int r = 0; r < 8; ++r) {
      float v = acc[nt][r] + bv;
      v = v > 0.f ? v : 0.f;
      act0[(m0 + hi * 8 + r) * 128 + nt * 16 + ln] = (_Float16)v;
    }
  }
  __syncthreads();

  // -------------------- layers 1..3: A and B both from LDS --------------------
  const _Float16* Wts[3] = { Wt1, Wt2, Wt3 };
  const float*    Bst[3] = { B1, B2, B3 };

  for (int L = 0; L < 3; ++L) {
    // stage the full 128x128 f16 weight (32KB) into LDS, shared by all waves
    const _Float16* Wl = Wts[L];
#pragma unroll
    for (int i = 0; i < 4; ++i) {
      int idx = tid * 64 + i * 16;
      *(v16h*)(bw + idx) = *(const v16h*)(Wl + idx);
    }
    __syncthreads();

    _Float16* rd = (L & 1) ? act1 : act0;
    _Float16* wr = (L & 1) ? act0 : act1;

#pragma unroll
    for (int t = 0; t < 8; ++t) acc[t] = zero8;

#pragma unroll
    for (int kk = 0; kk < 128; kk += 32) {
      v16h a;
      const _Float16* p = rd + arow * 128 + kk + hi * 8;
      v8h lo = *(const v8h*)(p);
      v8h h2 = *(const v8h*)(p + 16);
#pragma unroll
      for (int i = 0; i < 8; ++i) { a[i] = lo[i]; a[8 + i] = h2[i]; }

#pragma unroll
      for (int nt = 0; nt < 8; ++nt) {
        const _Float16* bp = bw + (nt * 16 + ln) * 128 + kk + hi * 8;
        v8h b0 = *(const v8h*)(bp);
        v8h b1 = *(const v8h*)(bp + 16);
        v16h b;
#pragma unroll
        for (int i = 0; i < 8; ++i) { b[i] = b0[i]; b[8 + i] = b1[i]; }
        acc[nt] = __builtin_amdgcn_wmma_f32_16x16x32_f16(
            false, a, false, b, (short)0, acc[nt], false, false);
      }
    }

    if (L < 2) {
      const float* Bl = Bst[L];
#pragma unroll
      for (int nt = 0; nt < 8; ++nt) {
        float bv = Bl[nt * 16 + ln];
#pragma unroll
        for (int r = 0; r < 8; ++r) {
          float v = acc[nt][r] + bv;
          v = v > 0.f ? v : 0.f;
          wr[(m0 + hi * 8 + r) * 128 + nt * 16 + ln] = (_Float16)v;
        }
      }
      __syncthreads();   // also orders bw re-staging for the next layer
    }
  }

  // -------------------- final epilogue: bias, stores, fused scatter-sum ------
  int  dsts[8];
  bool vrow[8];
#pragma unroll
  for (int r = 0; r < 8; ++r) {
    long g = rb + m0 + hi * 8 + r;
    vrow[r] = (g < R);
    dsts[r] = (agg_dst != nullptr && vrow[r]) ? agg_dst[g] : 0;
  }
#pragma unroll
  for (int nt = 0; nt < 8; ++nt) {
    float bv = B3[nt * 16 + ln];
#pragma unroll
    for (int r = 0; r < 8; ++r) {
      if (!vrow[r]) continue;
      long g = rb + m0 + hi * 8 + r;
      float v = acc[nt][r] + bv;
      long o = g * 128 + nt * 16 + ln;
      if (out_h) out_h[o] = (_Float16)v;
      if (out_f) out_f[o] = v;
      if (agg_sum) atomicAdd(agg_sum + (long)dsts[r] * 128 + nt * 16 + ln, v);
    }
  }
}

// ---------------------------------------------------------------------------
// Small helper kernels
// ---------------------------------------------------------------------------
__global__ void k_f32_to_f16(const float* __restrict__ x, _Float16* __restrict__ y, long n) {
  long i = (long)blockIdx.x * blockDim.x + threadIdx.x;
  if (i < n) y[i] = (_Float16)x[i];
}

// W (K x 128, row-major f32) -> Wt (128 x Kpad, f16, zero padded beyond K)
__global__ void k_convert_wt(const float* __restrict__ W, _Float16* __restrict__ Wt,
                             int K, int Kpad, long total) {
  long i = (long)blockIdx.x * blockDim.x + threadIdx.x;
  if (i >= total) return;
  int n = (int)(i / Kpad);
  int k = (int)(i % Kpad);
  Wt[i] = (k < K) ? (_Float16)W[(long)k * 128 + n] : (_Float16)0.f;
}

__global__ void k_zero_f32(float* __restrict__ p, long n) {
  long i = (long)blockIdx.x * blockDim.x + threadIdx.x;
  if (i < n) p[i] = 0.f;
}

__global__ void k_count(const int* __restrict__ dst, int E, float* __restrict__ cnt) {
  int i = blockIdx.x * blockDim.x + threadIdx.x;
  if (i < E) atomicAdd(cnt + dst[i], 1.0f);
}

// out[n,c] = (f16)(sum[n,c] / max(cnt[n],1))
__global__ void k_mean_h(const float* __restrict__ sum, const float* __restrict__ cnt,
                         _Float16* __restrict__ out, long total) {
  long i = (long)blockIdx.x * blockDim.x + threadIdx.x;
  if (i < total) out[i] = (_Float16)(sum[i] / fmaxf(cnt[i >> 7], 1.0f));
}

// column sums of an (R x 128) f16 matrix; 128 threads/block, 512 rows/block
__global__ void k_colsum(const _Float16* __restrict__ x, int R, float* __restrict__ cs) {
  int c  = threadIdx.x;
  int r0 = blockIdx.x * 512;
  int r1 = r0 + 512; if (r1 > R) r1 = R;
  float s = 0.f;
  for (int r = r0; r < r1; ++r) s += (float)x[(long)r * 128 + c];
  atomicAdd(cs + c, s);
}

// assemble global-MLP input: [ua(na) | ub(nb) | mean_xt | mean_xp | mean_ett | mean_etp]
__global__ void k_build_gin(float* __restrict__ gin,
                            const float* ua, int na, const float* ub, int nb,
                            const float* cs_xt, float inv_t,
                            const float* cs_xp, float inv_p,
                            const float* cs_ett, float inv_tt,
                            const float* cs_etp, float inv_tp) {
  int t = threadIdx.x;
  for (int i = t; i < na; i += 256) gin[i] = ua[i];
  for (int i = t; i < nb; i += 256) gin[na + i] = ub[i];
  int base = na + nb;
  for (int i = t; i < 128; i += 256) {
    gin[base + i]       = cs_xt[i]  * inv_t;
    gin[base + 128 + i] = cs_xp[i]  * inv_p;
    gin[base + 256 + i] = cs_ett[i] * inv_tt;
    gin[base + 384 + i] = cs_etp[i] * inv_tp;
  }
}

// 1-row 4-layer MLP (global updates), f32 VALU, 128 threads / 1 block
__global__ __launch_bounds__(128) void k_mlp4_single(
    const float* __restrict__ x, int K0,
    const float* W0, const float* b0, const float* W1, const float* b1,
    const float* W2, const float* b2, const float* W3, const float* b3,
    float* __restrict__ out_f, _Float16* __restrict__ out_h)
{
  __shared__ float xb[1024];
  __shared__ float yb[128];
  int t = threadIdx.x;
  for (int k = t; k < K0; k += 128) xb[k] = x[k];
  __syncthreads();
  float a = b0[t];
  for (int k = 0; k < K0; ++k) a += xb[k] * W0[(long)k * 128 + t];
  a = fmaxf(a, 0.f);
  yb[t] = a; __syncthreads(); xb[t] = yb[t]; __syncthreads();
  a = b1[t];
  for (int k = 0; k < 128; ++k) a += xb[k] * W1[k * 128 + t];
  a = fmaxf(a, 0.f);
  yb[t] = a; __syncthreads(); xb[t] = yb[t]; __syncthreads();
  a = b2[t];
  for (int k = 0; k < 128; ++k) a += xb[k] * W2[k * 128 + t];
  a = fmaxf(a, 0.f);
  yb[t] = a; __syncthreads(); xb[t] = yb[t]; __syncthreads();
  a = b3[t];
  for (int k = 0; k < 128; ++k) a += xb[k] * W3[k * 128 + t];
  if (out_f) out_f[t] = a;
  if (out_h) out_h[t] = (_Float16)a;
}

// per-edge linear head: out[e, :C] = x[e, :128] @ W(128xC) + b
__global__ void k_head(const _Float16* __restrict__ x, int R, int C,
                       const float* __restrict__ W, const float* __restrict__ b,
                       float* __restrict__ out) {
  int e = blockIdx.x * blockDim.x + threadIdx.x;
  if (e >= R) return;
  for (int j = 0; j < C; ++j) {
    float acc = b[j];
    for (int k = 0; k < 128; ++k) acc += (float)x[(long)e * 128 + k] * W[k * C + j];
    out[(long)e * C + j] = acc;
  }
}

// ---------------------------------------------------------------------------
// Host orchestration
// ---------------------------------------------------------------------------
static inline long cdiv(long a, long b) { return (a + b - 1) / b; }

extern "C" void kernel_launch(void* const* d_in, const int* in_sizes, int n_in,
                              void* d_out, int out_size, void* d_ws, size_t ws_size,
                              hipStream_t stream) {
  (void)in_sizes; (void)n_in; (void)out_size; (void)ws_size;

  const int NT = 10000, NP = 1000, ETT = 160000, ETP = 50000;
  const int FN = 16, FE = 8, FG = 8;

  const float* x_tracks = (const float*)d_in[0];
  const float* x_pvs    = (const float*)d_in[1];
  const float* e_tt     = (const float*)d_in[2];
  const float* e_tp     = (const float*)d_in[3];
  const float* u_in     = (const float*)d_in[4];
  const int*   src_tt   = (const int*)d_in[5];
  const int*   dst_tt   = (const int*)d_in[6];
  const int*   src_tp   = (const int*)d_in[7];
  const int*   dst_tp   = (const int*)d_in[8];

  // params: jax pytree flatten, dict keys sorted alphabetically:
  //   block0, block1, dec, enc, out_tt, out_tp
  //   within each block/coder: edge_tp, edge_tt, global, node_pvs, node_tracks
  //   each MLP: [(W0,b0),(W1,b1),(W2,b2),(W3,b3)] -> 8 arrays
  enum {
    B0_ETP = 9,   B0_ETT = 17,  B0_GLB = 25,  B0_NPV = 33,  B0_NTR = 41,
    B1_ETP = 49,  B1_ETT = 57,  B1_GLB = 65,  B1_NPV = 73,  B1_NTR = 81,
    DE_ETP = 89,  DE_ETT = 97,  DE_GLB = 105, DE_NPV = 113, DE_NTR = 121,
    EN_ETP = 129, EN_ETT = 137, EN_GLB = 145, EN_NPV = 153, EN_NTR = 161,
    OTT_W = 169,  OTT_B = 170,  OTP_W = 171,  OTP_B = 172
  };
  auto P = [&](int i) { return (const float*)d_in[i]; };

  // bump allocator over workspace
  char* wsb = (char*)d_ws;
  size_t off = 0;
  auto alloc = [&](size_t bytes) -> void* {
    off = (off + 255) & ~(size_t)255;
    void* p = wsb + off;
    off += bytes;
    return p;
  };

  // f16 copies of raw inputs
  _Float16* xtr_h = (_Float16*)alloc((size_t)NT * FN * 2);
  _Float16* xpv_h = (_Float16*)alloc((size_t)NP * FN * 2);
  _Float16* ett_h = (_Float16*)alloc((size_t)ETT * FE * 2);
  _Float16* etp_h = (_Float16*)alloc((size_t)ETP * FE * 2);

  // activations (all 128-wide, f16)
  _Float16* xt0  = (_Float16*)alloc((size_t)NT * 128 * 2);
  _Float16* xp0  = (_Float16*)alloc((size_t)NP * 128 * 2);
  _Float16* ett0 = (_Float16*)alloc((size_t)ETT * 128 * 2);
  _Float16* etp0 = (_Float16*)alloc((size_t)ETP * 128 * 2);
  _Float16* xt1  = (_Float16*)alloc((size_t)NT * 128 * 2);
  _Float16* xp1  = (_Float16*)alloc((size_t)NP * 128 * 2);
  _Float16* ett1 = (_Float16*)alloc((size_t)ETT * 128 * 2);
  _Float16* etp1 = (_Float16*)alloc((size_t)ETP * 128 * 2);
  _Float16* xt2  = (_Float16*)alloc((size_t)NT * 128 * 2);
  _Float16* xp2  = (_Float16*)alloc((size_t)NP * 128 * 2);
  _Float16* ett2 = (_Float16*)alloc((size_t)ETT * 128 * 2);
  _Float16* etp2 = (_Float16*)alloc((size_t)ETP * 128 * 2);
  _Float16* ett_d = ett0;   // reuse: ett0/etp0 dead after block1 edge kernels
  _Float16* etp_d = etp0;

  float*    u0_f = (float*)alloc(128 * 4);
  _Float16* u0_h = (_Float16*)alloc(128 * 2);
  float*    u1_f = (float*)alloc(128 * 4);
  _Float16* u1_h = (_Float16*)alloc(128 * 2);
  float*    u2_f = (float*)alloc(128 * 4);

  // scatter-mean buffers
  float*    aggs_t = (float*)alloc((size_t)NT * 128 * 4);
  float*    aggs_p = (float*)alloc((size_t)NP * 128 * 4);
  _Float16* aggm_t = (_Float16*)alloc((size_t)NT * 128 * 2);
  _Float16* aggm_p = (_Float16*)alloc((size_t)NP * 128 * 2);
  float*    cnt_t  = (float*)alloc((size_t)NT * 4);
  float*    cnt_p  = (float*)alloc((size_t)NP * 4);
  float*    cs     = (float*)alloc(512 * 4);
  float*    gin    = (float*)alloc(1024 * 4);

  // weight conversion (f32 row-major -> f16 transposed [128][Kpad])
  struct WtSet { const _Float16* Wt[4]; const float* B[4]; int K0; };
  auto convert_mlp = [&](int base, int Kin, int Kpad) -> WtSet {
    WtSet s; s.K0 = Kpad;
    int Ks[4]  = {Kin, 128, 128, 128};
    int Kps[4] = {Kpad, 128, 128, 128};
    for (int l = 0; l < 4; ++l) {
      long total = (long)128 * Kps[l];
      _Float16* wt = (_Float16*)alloc((size_t)total * 2);
      k_convert_wt<<<dim3((unsigned)cdiv(total, 256)), dim3(256), 0, stream>>>(
          P(base + 2 * l), wt, Ks[l], Kps[l], total);
      s.Wt[l] = wt;
      s.B[l]  = P(base + 2 * l + 1);
    }
    return s;
  };

  WtSet en_ntr = convert_mlp(EN_NTR, 16, 32);
  WtSet en_npv = convert_mlp(EN_NPV, 16, 32);
  WtSet en_ett = convert_mlp(EN_ETT, 8, 32);
  WtSet en_etp = convert_mlp(EN_ETP, 8, 32);
  WtSet b0_ett = convert_mlp(B0_ETT, 512, 512);
  WtSet b0_etp = convert_mlp(B0_ETP, 512, 512);
  WtSet b0_ntr = convert_mlp(B0_NTR, 384, 384);
  WtSet b0_npv = convert_mlp(B0_NPV, 384, 384);
  WtSet b1_ett = convert_mlp(B1_ETT, 1024, 1024);
  WtSet b1_etp = convert_mlp(B1_ETP, 1024, 1024);
  WtSet b1_ntr = convert_mlp(B1_NTR, 640, 640);
  WtSet b1_npv = convert_mlp(B1_NPV, 640, 640);
  WtSet de_ntr = convert_mlp(DE_NTR, 128, 128);
  WtSet de_npv = convert_mlp(DE_NPV, 128, 128);
  WtSet de_ett = convert_mlp(DE_ETT, 128, 128);
  WtSet de_etp = convert_mlp(DE_ETP, 128, 128);

  // raw input conversions
  k_f32_to_f16<<<dim3((unsigned)cdiv((long)NT * FN, 256)), 256, 0, stream>>>(x_tracks, xtr_h, (long)NT * FN);
  k_f32_to_f16<<<dim3((unsigned)cdiv((long)NP * FN, 256)), 256, 0, stream>>>(x_pvs, xpv_h, (long)NP * FN);
  k_f32_to_f16<<<dim3((unsigned)cdiv((long)ETT * FE, 256)), 256, 0, stream>>>(e_tt, ett_h, (long)ETT * FE);
  k_f32_to_f16<<<dim3((unsigned)cdiv((long)ETP * FE, 256)), 256, 0, stream>>>(e_tp, etp_h, (long)ETP * FE);

  // segment-mean denominators (same for both blocks)
  k_zero_f32<<<dim3((unsigned)cdiv(NT, 256)), 256, 0, stream>>>(cnt_t, NT);
  k_zero_f32<<<dim3((unsigned)cdiv(NP, 256)), 256, 0, stream>>>(cnt_p, NP);
  k_count<<<dim3((unsigned)cdiv(ETT, 256)), 256, 0, stream>>>(dst_tt, ETT, cnt_t);
  k_count<<<dim3((unsigned)cdiv(ETP, 256)), 256, 0, stream>>>(dst_tp, ETP, cnt_p);

  auto mkseg = [](const _Float16* base, const int* idx, int stride, int vw, int wpad, int mode) {
    SegDesc d; d.base = base; d.idx = idx; d.stride = stride; d.start = 0;
    d.vw = vw; d.wpad = wpad; d.mode = mode; return d;
  };
  auto mksegs = [](std::initializer_list<SegDesc> l) {
    Segs s; int i = 0, st = 0;
    for (const SegDesc& d : l) { s.s[i] = d; s.s[i].start = st; st += d.wpad; ++i; }
    s.n = i;
    for (; i < 8; ++i) { s.s[i] = SegDesc{nullptr, nullptr, 0, 0, 0, 0, 0}; }
    return s;
  };
  auto run_mlp = [&](int R, const Segs& sg, const WtSet& w,
                     _Float16* oh, float* of, float* asum, const int* adst) {
    k_fused_mlp4<<<dim3((unsigned)cdiv(R, 128)), dim3(256), 0, stream>>>(
        R, sg, w.K0, w.Wt[0], w.B[0], w.Wt[1], w.B[1], w.Wt[2], w.B[2], w.Wt[3], w.B[3],
        oh, of, asum, adst);
  };

  // ---------------- encoder ----------------
  run_mlp(NT,  mksegs({mkseg(xtr_h, nullptr, FN, FN, 32, 0)}), en_ntr, xt0, nullptr, nullptr, nullptr);
  run_mlp(NP,  mksegs({mkseg(xpv_h, nullptr, FN, FN, 32, 0)}), en_npv, xp0, nullptr, nullptr, nullptr);
  run_mlp(ETT, mksegs({mkseg(ett_h, nullptr, FE, FE, 32, 0)}), en_ett, ett0, nullptr, nullptr, nullptr);
  run_mlp(ETP, mksegs({mkseg(etp_h, nullptr, FE, FE, 32, 0)}), en_etp, etp0, nullptr, nullptr, nullptr);
  k_mlp4_single<<<1, 128, 0, stream>>>(u_in, FG,
      P(EN_GLB + 0), P(EN_GLB + 1), P(EN_GLB + 2), P(EN_GLB + 3),
      P(EN_GLB + 4), P(EN_GLB + 5), P(EN_GLB + 6), P(EN_GLB + 7), u0_f, u0_h);

  // ---------------- block 0 (F = 128) ----------------
  k_zero_f32<<<dim3((unsigned)cdiv((long)NT * 128, 256)), 256, 0, stream>>>(aggs_t, (long)NT * 128);
  k_zero_f32<<<dim3((unsigned)cdiv((long)NP * 128, 256)), 256, 0, stream>>>(aggs_p, (long)NP * 128);
  run_mlp(ETT, mksegs({mkseg(ett0, nullptr, 128, 128, 128, 0),
                       mkseg(xt0,  src_tt,  128, 128, 128, 1),
                       mkseg(xt0,  dst_tt,  128, 128, 128, 1),
                       mkseg(u0_h, nullptr, 128, 128, 128, 2)}),
          b0_ett, ett1, nullptr, aggs_t, dst_tt);
  run_mlp(ETP, mksegs({mkseg(etp0, nullptr, 128, 128, 128, 0),
                       mkseg(xt0,  src_tp,  128, 128, 128, 1),
                       mkseg(xp0,  dst_tp,  128, 128, 128, 1),
                       mkseg(u0_h, nullptr, 128, 128, 128, 2)}),
          b0_etp, etp1, nullptr, aggs_p, dst_tp);
  k_mean_h<<<dim3((unsigned)cdiv((long)NT * 128, 256)), 256, 0, stream>>>(aggs_t, cnt_t, aggm_t, (long)NT * 128);
  k_mean_h<<<dim3((unsigned)cdiv((long)NP * 128, 256)), 256, 0, stream>>>(aggs_p, cnt_p, aggm_p, (long)NP * 128);
  run_mlp(NT, mksegs({mkseg(xt0,    nullptr, 128, 128, 128, 0),
                      mkseg(aggm_t, nullptr, 128, 128, 128, 0),
                      mkseg(u0_h,   nullptr, 128, 128, 128, 2)}),
          b0_ntr, xt1, nullptr, nullptr, nullptr);
  run_mlp(NP, mksegs({mkseg(xp0,    nullptr, 128, 128, 128, 0),
                      mkseg(aggm_p, nullptr, 128, 128, 128, 0),
                      mkseg(u0_h,   nullptr, 128, 128, 128, 2)}),
          b0_npv, xp1, nullptr, nullptr, nullptr);
  k_zero_f32<<<2, 256, 0, stream>>>(cs, 512);
  k_colsum<<<dim3((unsigned)cdiv(NT, 512)), 128, 0, stream>>>(xt1, NT, cs + 0);
  k_colsum<<<dim3((unsigned)cdiv(NP, 512)), 128, 0, stream>>>(xp1, NP, cs + 128);
  k_colsum<<<dim3((unsigned)cdiv(ETT, 512)), 128, 0, stream>>>(ett1, ETT, cs + 256);
  k_colsum<<<dim3((unsigned)cdiv(ETP, 512)), 128, 0, stream>>>(etp1, ETP, cs + 384);
  k_build_gin<<<1, 256, 0, stream>>>(gin, u0_f, 128, nullptr, 0,
      cs + 0, 1.f / NT, cs + 128, 1.f / NP, cs + 256, 1.f / ETT, cs + 384, 1.f / ETP);
  k_mlp4_single<<<1, 128, 0, stream>>>(gin, 640,
      P(B0_GLB + 0), P(B0_GLB + 1), P(B0_GLB + 2), P(B0_GLB + 3),
      P(B0_GLB + 4), P(B0_GLB + 5), P(B0_GLB + 6), P(B0_GLB + 7), u1_f, u1_h);

  // ---------------- block 1 (F = 256, skip concat handled as extra segments) --
  k_zero_f32<<<dim3((unsigned)cdiv((long)NT * 128, 256)), 256, 0, stream>>>(aggs_t, (long)NT * 128);
  k_zero_f32<<<dim3((unsigned)cdiv((long)NP * 128, 256)), 256, 0, stream>>>(aggs_p, (long)NP * 128);
  run_mlp(ETT, mksegs({mkseg(ett0, nullptr, 128, 128, 128, 0),
                       mkseg(ett1, nullptr, 128, 128, 128, 0),
                       mkseg(xt0,  src_tt,  128, 128, 128, 1),
                       mkseg(xt1,  src_tt,  128, 128, 128, 1),
                       mkseg(xt0,  dst_tt,  128, 128, 128, 1),
                       mkseg(xt1,  dst_tt,  128, 128, 128, 1),
                       mkseg(u0_h, nullptr, 128, 128, 128, 2),
                       mkseg(u1_h, nullptr, 128, 128, 128, 2)}),
          b1_ett, ett2, nullptr, aggs_t, dst_tt);
  run_mlp(ETP, mksegs({mkseg(etp0, nullptr, 128, 128, 128, 0),
                       mkseg(etp1, nullptr, 128, 128, 128, 0),
                       mkseg(xt0,  src_tp,  128, 128, 128, 1),
                       mkseg(xt1,  src_tp,  128, 128, 128, 1),
                       mkseg(xp0,  dst_tp,  128, 128, 128, 1),
                       mkseg(xp1,  dst_tp,  128, 128, 128, 1),
                       mkseg(u0_h, nullptr, 128, 128, 128, 2),
                       mkseg(u1_h, nullptr, 128, 128, 128, 2)}),
          b1_etp, etp2, nullptr, aggs_p, dst_tp);
  k_mean_h<<<dim3((unsigned)cdiv((long)NT * 128, 256)), 256, 0, stream>>>(aggs_t, cnt_t, aggm_t, (long)NT * 128);
  k_mean_h<<<dim3((unsigned)cdiv((long)NP * 128, 256)), 256, 0, stream>>>(aggs_p, cnt_p, aggm_p, (long)NP * 128);
  run_mlp(NT, mksegs({mkseg(xt0,    nullptr, 128, 128, 128, 0),
                      mkseg(xt1,    nullptr, 128, 128, 128, 0),
                      mkseg(aggm_t, nullptr, 128, 128, 128, 0),
                      mkseg(u0_h,   nullptr, 128, 128, 128, 2),
                      mkseg(u1_h,   nullptr, 128, 128, 128, 2)}),
          b1_ntr, xt2, nullptr, nullptr, nullptr);
  run_mlp(NP, mksegs({mkseg(xp0,    nullptr, 128, 128, 128, 0),
                      mkseg(xp1,    nullptr, 128, 128, 128, 0),
                      mkseg(aggm_p, nullptr, 128, 128, 128, 0),
                      mkseg(u0_h,   nullptr, 128, 128, 128, 2),
                      mkseg(u1_h,   nullptr, 128, 128, 128, 2)}),
          b1_npv, xp2, nullptr, nullptr, nullptr);
  k_zero_f32<<<2, 256, 0, stream>>>(cs, 512);
  k_colsum<<<dim3((unsigned)cdiv(NT, 512)), 128, 0, stream>>>(xt2, NT, cs + 0);
  k_colsum<<<dim3((unsigned)cdiv(NP, 512)), 128, 0, stream>>>(xp2, NP, cs + 128);
  k_colsum<<<dim3((unsigned)cdiv(ETT, 512)), 128, 0, stream>>>(ett2, ETT, cs + 256);
  k_colsum<<<dim3((unsigned)cdiv(ETP, 512)), 128, 0, stream>>>(etp2, ETP, cs + 384);
  k_build_gin<<<1, 256, 0, stream>>>(gin, u0_f, 128, u1_f, 128,
      cs + 0, 1.f / NT, cs + 128, 1.f / NP, cs + 256, 1.f / ETT, cs + 384, 1.f / ETP);
  k_mlp4_single<<<1, 128, 0, stream>>>(gin, 768,
      P(B1_GLB + 0), P(B1_GLB + 1), P(B1_GLB + 2), P(B1_GLB + 3),
      P(B1_GLB + 4), P(B1_GLB + 5), P(B1_GLB + 6), P(B1_GLB + 7), u2_f, nullptr);

  // ---------------- decoder + output heads ----------------
  float* out_xt  = (float*)d_out;
  float* out_xp  = out_xt + (long)NT * 128;
  float* out_ett = out_xp + (long)NP * 128;
  float* out_etp = out_ett + (long)ETT * 4;
  float* out_u   = out_etp + (long)ETP * 1;

  run_mlp(NT,  mksegs({mkseg(xt2,  nullptr, 128, 128, 128, 0)}), de_ntr, nullptr, out_xt, nullptr, nullptr);
  run_mlp(NP,  mksegs({mkseg(xp2,  nullptr, 128, 128, 128, 0)}), de_npv, nullptr, out_xp, nullptr, nullptr);
  run_mlp(ETT, mksegs({mkseg(ett2, nullptr, 128, 128, 128, 0)}), de_ett, ett_d, nullptr, nullptr, nullptr);
  run_mlp(ETP, mksegs({mkseg(etp2, nullptr, 128, 128, 128, 0)}), de_etp, etp_d, nullptr, nullptr, nullptr);
  k_mlp4_single<<<1, 128, 0, stream>>>(u2_f, 128,
      P(DE_GLB + 0), P(DE_GLB + 1), P(DE_GLB + 2), P(DE_GLB + 3),
      P(DE_GLB + 4), P(DE_GLB + 5), P(DE_GLB + 6), P(DE_GLB + 7), out_u, nullptr);

  k_head<<<dim3((unsigned)cdiv(ETT, 256)), 256, 0, stream>>>(ett_d, ETT, 4, P(OTT_W), P(OTT_B), out_ett);
  k_head<<<dim3((unsigned)cdiv(ETP, 256)), 256, 0, stream>>>(etp_d, ETP, 1, P(OTP_W), P(OTP_B), out_etp);
}
